// Int8Linear_1803886265620
// MI455X (gfx1250) — matrix operand order repaired and checked by reading, not compiled, not run on hardware
//
#include <hip/hip_runtime.h>

typedef __attribute__((ext_vector_type(16))) __bf16 v16bf;
typedef __attribute__((ext_vector_type(8)))  float  v8f;

#define IN_F   4096
#define OUT_F  11008
#define M_TOT  8192
#define BM     128
#define BN     128
#define BK     64
#define LDSW   72            // padded LDS row stride (bf16 elems), 144B, 16B aligned
#define NIT    (IN_F / BK)   // 64 K-steps

// Double-buffered staging tiles; the output bounce tile aliases them after the K loop.
union SharedMem {
  struct { __bf16 x[2][BM * LDSW]; __bf16 w[2][BN * LDSW]; } stage;
  __bf16 out[BM * BN];
};

__device__ __forceinline__ void dequant_store(__bf16* dst, uint4 q0, uint4 q1) {
  signed char raw[32];
  *reinterpret_cast<uint4*>(&raw[0])  = q0;
  *reinterpret_cast<uint4*>(&raw[16]) = q1;
  #pragma unroll
  for (int i = 0; i < 32; ++i) dst[i] = (__bf16)(float)raw[i];
}

__device__ __forceinline__ void async_x_tile(unsigned lds_addr, const __bf16* gaddr) {
  // IOFFSET is added to BOTH the LDS and global addresses (ISA 15.18.3), so one
  // address pair covers the thread's 32B chunk with two b128 async transfers.
  asm volatile("global_load_async_to_lds_b128 %0, %1, off"
               :: "v"(lds_addr), "v"(gaddr) : "memory");
  asm volatile("global_load_async_to_lds_b128 %0, %1, off offset:16"
               :: "v"(lds_addr), "v"(gaddr) : "memory");
}

__global__ __launch_bounds__(256)
void int8_linear_wmma(const __bf16* __restrict__ X,
                      const signed char* __restrict__ W,
                      const __bf16* __restrict__ S,
                      __bf16* __restrict__ Y) {
  __shared__ SharedMem smem;

  const int t     = threadIdx.x;
  const int lane  = t & 31;
  const int wave  = t >> 5;
  const int waveM = wave >> 2;   // 0..1
  const int waveN = wave & 3;    // 0..3

  const int n0 = blockIdx.x * BN;
  const int m0 = blockIdx.y * BM;

  // staging coords: 2 threads per row, 32B per thread
  const int srow  = t >> 1;      // 0..127
  const int shalf = t & 1;       // 0..1
  const int soff  = srow * LDSW + shalf * 32;

  const __bf16*      xga = X + (size_t)(m0 + srow) * IN_F + shalf * 32;
  const signed char* wga = W + (size_t)(n0 + srow) * IN_F + shalf * 32;

  // fragment coords
  const int lrow = lane & 15;
  const int lhi  = lane >> 4;

  union Frag { uint4 q[2]; v16bf v; };

  v8f zero = {};
  v8f acc[4][2];
  #pragma unroll
  for (int i = 0; i < 4; ++i)
    #pragma unroll
    for (int j = 0; j < 2; ++j)
      acc[i][j] = zero;

  // ---- prologue: stage buffer 0
  {
    async_x_tile((unsigned)(size_t)&smem.stage.x[0][soff], xga);
    const uint4* wsrc = reinterpret_cast<const uint4*>(wga);
    uint4 q0 = wsrc[0];
    uint4 q1 = wsrc[1];
    dequant_store(&smem.stage.w[0][soff], q0, q1);
    asm volatile("s_wait_asynccnt 0x0" ::: "memory");
  }
  __syncthreads();

  for (int it = 0; it < NIT; ++it) {
    const int  buf  = it & 1;
    const bool more = (it + 1) < NIT;

    // ---- issue next tile's loads before computing (latency hiding)
    uint4 q0, q1;
    if (more) {
      async_x_tile((unsigned)(size_t)&smem.stage.x[buf ^ 1][soff],
                   xga + (size_t)(it + 1) * BK);
      const uint4* wsrc =
          reinterpret_cast<const uint4*>(wga + (size_t)(it + 1) * BK);
      q0 = wsrc[0];
      q1 = wsrc[1];
      if (it + 2 < NIT)
        __builtin_prefetch(wga + (size_t)(it + 2) * BK, 0, 2);
    }

    // ---- compute on current buffer
    const __bf16* lXb = smem.stage.x[buf];
    const __bf16* lWb = smem.stage.w[buf];
    #pragma unroll
    for (int kk = 0; kk < BK; kk += 32) {
      Frag a[4], b[2];
      #pragma unroll
      for (int am = 0; am < 4; ++am) {
        const int row = waveM * 64 + am * 16 + lrow;
        a[am].q[0] = *reinterpret_cast<const uint4*>(&lXb[row * LDSW + kk + lhi * 8]);
        a[am].q[1] = *reinterpret_cast<const uint4*>(&lXb[row * LDSW + kk + 16 + lhi * 8]);
      }
      #pragma unroll
      for (int bn = 0; bn < 2; ++bn) {
        const int row = waveN * 32 + bn * 16 + lrow;
        b[bn].q[0] = *reinterpret_cast<const uint4*>(&lWb[row * LDSW + kk + lhi * 8]);
        b[bn].q[1] = *reinterpret_cast<const uint4*>(&lWb[row * LDSW + kk + 16 + lhi * 8]);
      }
      #pragma unroll
      for (int am = 0; am < 4; ++am)
        #pragma unroll
        for (int bn = 0; bn < 2; ++bn)
          acc[am][bn] = __builtin_amdgcn_wmma_f32_16x16x32_bf16(
              false, a[am].v, false, b[bn].v, (short)0, acc[am][bn], false, false);
    }

    // ---- finish staging the next buffer behind the WMMAs
    if (more) {
      dequant_store(&smem.stage.w[buf ^ 1][soff], q0, q1);
      asm volatile("s_wait_asynccnt 0x0" ::: "memory");
    }
    __syncthreads();
  }

  // ---- epilogue: scale in f32, pack bf16 into LDS tile (aliases staging bufs)
  #pragma unroll
  for (int bn = 0; bn < 2; ++bn) {
    const int ncol = waveN * 32 + bn * 16 + lrow;
    const float s = (float)S[n0 + ncol];
    #pragma unroll
    for (int am = 0; am < 4; ++am) {
      const int mbase = waveM * 64 + am * 16 + lhi * 8;
      #pragma unroll
      for (int r = 0; r < 8; ++r)
        smem.out[(mbase + r) * BN + ncol] = (__bf16)(acc[am][bn][r] * s);
    }
  }
  __syncthreads();

  // ---- coalesced writeback: 128B contiguous per thread, b128 stores
  {
    const int orow  = t >> 1;
    const int ohalf = t & 1;
    const uint4* src = reinterpret_cast<const uint4*>(&smem.out[orow * BN + ohalf * 64]);
    uint4* dst = reinterpret_cast<uint4*>(Y + (size_t)(m0 + orow) * OUT_F + n0 + ohalf * 64);
    #pragma unroll
    for (int i = 0; i < 8; ++i) dst[i] = src[i];
  }
}

extern "C" void kernel_launch(void* const* d_in, const int* in_sizes, int n_in,
                              void* d_out, int out_size, void* d_ws, size_t ws_size,
                              hipStream_t stream) {
  const __bf16*      x = (const __bf16*)d_in[0];
  const signed char* w = (const signed char*)d_in[1];
  const __bf16*      s = (const __bf16*)d_in[2];
  __bf16*            y = (__bf16*)d_out;

  dim3 grid(OUT_F / BN, M_TOT / BM, 1);   // 86 x 64, exact tiling
  dim3 block(256, 1, 1);                  // 8 wave32 waves
  hipLaunchKernelGGL(int8_linear_wmma, grid, block, 0, stream, x, w, s, y);
}
